// BF_Attention_3169685864936
// MI455X (gfx1250) — compile-verified, matrix operand order checked by reading, not run on hardware
//
#include <hip/hip_runtime.h>
#include <hip/hip_bf16.h>
#include <cstdint>
#include <cstddef>

// ---------------------------------------------------------------------------
// BF_Attention for MI455X (gfx1250, wave32, WMMA)
//
// Only ONE large GEMM survives the algebraic collapse:
//   Y[n] = Waug[n] (272x256, bf16) x X[n] (256x9216)  via v_wmma_f32_16x16x32_bf16
// Waug rows 0..255 = W_f, rows 256/257 = u_k = W_v^T bf_feat_k (attention logit
// vectors), rows 258..271 = 0 padding.
//
// Main kernel: 8 waves, pixel tile 256; each wave owns TWO 16-pixel N-tiles and
// all 17 M-tiles -> every A fragment feeds 2 WMMAs (halves redundant A traffic).
// ---------------------------------------------------------------------------

typedef __attribute__((ext_vector_type(16))) __bf16 v16bf;
typedef __attribute__((ext_vector_type(8)))  float  v8f;
typedef __attribute__((ext_vector_type(4)))  float  v4f;

#define C_DIM 256
#define N_DIM 16
#define HW 9216            // 96*96
#define PIX_TILE 256
#define O_AUG 272          // 256 + 16 (rows 256,257 = u vectors, rest zero)
#define LDS_STRIDE 264     // 256 + 8 halfword pad (bank-conflict mitigation)

union FragBF {             // trivially-constructible pun: 16 bf16 <-> 2 x 16B
  v16bf v;
  v4f   f[2];
};

// --- mask sums: msum[k*16+n] = 4 * sum(mask[n])  (nearest 2x upsample = x4) ---
__global__ __launch_bounds__(256) void k_msum(const float* __restrict__ bg,
                                              const float* __restrict__ fg,
                                              float* __restrict__ msum) {
  int bx = blockIdx.x;                 // 0..31
  int n = bx & 15, k = bx >> 4;
  const float* src = (k ? fg : bg) + n * 2304;
  float s = 0.f;
  for (int i = threadIdx.x; i < 2304; i += 256) s += src[i];
  for (int off = 16; off > 0; off >>= 1) s += __shfl_down(s, off, 32);
  __shared__ float red[8];
  int wave = threadIdx.x >> 5, lane = threadIdx.x & 31;
  if (lane == 0) red[wave] = s;
  __syncthreads();
  if (threadIdx.x == 0) {
    float t = 0.f;
    for (int w = 0; w < 8; ++w) t += red[w];
    msum[k * 16 + n] = 4.0f * t;
  }
}

// --- masked channel sums: xm[k][n*256+c] = sum_s mask_up[n,s] * x[n,c,s] ---
__global__ __launch_bounds__(256) void k_xm(const float* __restrict__ x,
                                            const float* __restrict__ bg,
                                            const float* __restrict__ fg,
                                            float* __restrict__ xm) {
  int bx = blockIdx.x;                 // n*256 + c  (4096 blocks)
  int n = bx >> 8;
  const float* xr  = x + (size_t)bx * HW;
  const float* bgm = bg + n * 2304;
  const float* fgm = fg + n * 2304;
  float sb = 0.f, sf = 0.f;
  for (int s = threadIdx.x; s < HW; s += 256) {
    int h = s / 96, w = s - h * 96;
    int mi = (h >> 1) * 48 + (w >> 1);
    float v = xr[s];
    sb += v * bgm[mi];
    sf += v * fgm[mi];
  }
  for (int off = 16; off > 0; off >>= 1) {
    sb += __shfl_down(sb, off, 32);
    sf += __shfl_down(sf, off, 32);
  }
  __shared__ float rb[8], rf[8];
  int wave = threadIdx.x >> 5, lane = threadIdx.x & 31;
  if (lane == 0) { rb[wave] = sb; rf[wave] = sf; }
  __syncthreads();
  if (threadIdx.x == 0) {
    float tb = 0.f, tf = 0.f;
    for (int w = 0; w < 8; ++w) { tb += rb[w]; tf += rf[w]; }
    xm[0 * 4096 + bx] = tb;
    xm[1 * 4096 + bx] = tf;
  }
}

// --- bf_feat: feat[k][n*256+o] = scale_k * (W_fb[o,:].xm_k[n,:] + b_fb[o]*msum_k[n]) ---
__global__ __launch_bounds__(256) void k_feat(const float* __restrict__ W_fb,
                                              const float* __restrict__ b_fb,
                                              const float* __restrict__ xm,
                                              const float* __restrict__ msum,
                                              float* __restrict__ feat) {
  int bx = blockIdx.x;                 // 0..31
  int n = bx >> 1, k = bx & 1;
  int o = threadIdx.x;
  float total = 0.f;
  for (int n2 = 0; n2 < 16; ++n2) total += msum[k * 16 + n2];
  float scale = (147456.0f / total) * (1.0f / 9216.0f);   // ratio * 1/(H*W)
  const float* xmr = xm + k * 4096 + n * 256;
  const float* wr  = W_fb + o * 256;
  float acc = 0.f;
  for (int c = 0; c < 256; ++c) acc += wr[c] * xmr[c];
  acc += b_fb[o] * msum[k * 16 + n];
  feat[k * 4096 + n * 256 + o] = scale * acc;
}

// --- u[k][n*256+c] = W_v^T . feat_k ; dvals[k*16+n] = b_v . feat_k ---
__global__ __launch_bounds__(256) void k_uvec(const float* __restrict__ W_v,
                                              const float* __restrict__ b_v,
                                              const float* __restrict__ feat,
                                              float* __restrict__ u,
                                              float* __restrict__ dvals) {
  int bx = blockIdx.x;
  int n = bx >> 1, k = bx & 1;
  int c = threadIdx.x;
  const float* fr = feat + k * 4096 + n * 256;
  float acc = 0.f;
  for (int o = 0; o < 256; ++o) acc += W_v[o * 256 + c] * fr[o];
  u[k * 4096 + n * 256 + c] = acc;
  float dv = b_v[c] * fr[c];
  for (int off = 16; off > 0; off >>= 1) dv += __shfl_down(dv, off, 32);
  __shared__ float red[8];
  int wave = threadIdx.x >> 5, lane = threadIdx.x & 31;
  if (lane == 0) red[wave] = dv;
  __syncthreads();
  if (threadIdx.x == 0) {
    float t = 0.f;
    for (int w = 0; w < 8; ++w) t += red[w];
    dvals[k * 16 + n] = t;
  }
}

// --- build per-batch augmented bf16 weight: [16][272][256] ---
__global__ __launch_bounds__(1024) void k_build_waug(const float* __restrict__ W_f,
                                                     const float* __restrict__ u,
                                                     __bf16* __restrict__ waug) {
  int idx = blockIdx.x * 1024 + threadIdx.x;   // 16*272*256 = 1,114,112 exactly
  int n = idx / (O_AUG * 256);
  int rem = idx - n * (O_AUG * 256);
  int r = rem >> 8, c = rem & 255;
  float v;
  if (r < 256)      v = W_f[r * 256 + c];
  else if (r < 258) v = u[(r - 256) * 4096 + n * 256 + c];
  else              v = 0.0f;
  waug[idx] = (__bf16)v;
}

// --- main fused GEMM + attention epilogue ------------------------------------
// grid (36, 16): blockIdx.x = 256-pixel tile, blockIdx.y = batch n.
// 8 waves; wave w owns pixels [s0+32w, s0+32w+32) as two N-tiles and all 17
// M-tiles. Each A fragment (Waug tile) is reused by 2 WMMAs.
__global__ __launch_bounds__(256) void k_main(const float* __restrict__ x,
                                              const __bf16* __restrict__ waug,
                                              const float* __restrict__ b_f,
                                              const float* __restrict__ feat,
                                              const float* __restrict__ dvals,
                                              const float* __restrict__ gamma,
                                              float* __restrict__ out) {
  const int n  = blockIdx.y;
  const int s0 = blockIdx.x * PIX_TILE;
  const int tid  = threadIdx.x;
  const int wave = tid >> 5, lane = tid & 31;
  const int mrow = lane & 15;
  const bool hi  = lane >= 16;

  __shared__ __align__(16) __bf16 xT[PIX_TILE * LDS_STRIDE];  // 135,168 B
  __shared__ float s_bf[256], s_bg[256], s_fg[256];

  // Stage x tile (fp32 global, coalesced b128 loads) -> bf16 transposed LDS [pix][c].
  const float* xn = x + (size_t)n * 256 * HW + s0;
  for (int i = tid; i < 256 * 64; i += 256) {
    int c  = i >> 6;
    int p4 = (i & 63) << 2;
    v4f v = *(const v4f*)(xn + (size_t)c * HW + p4);
    xT[(p4 + 0) * LDS_STRIDE + c] = (__bf16)v.x;
    xT[(p4 + 1) * LDS_STRIDE + c] = (__bf16)v.y;
    xT[(p4 + 2) * LDS_STRIDE + c] = (__bf16)v.z;
    xT[(p4 + 3) * LDS_STRIDE + c] = (__bf16)v.w;
  }
  s_bf[tid] = b_f[tid];
  s_bg[tid] = feat[0 * 4096 + n * 256 + tid];
  s_fg[tid] = feat[1 * 4096 + n * 256 + tid];
  __syncthreads();

  v8f acc0[17] = {};
  v8f acc1[17] = {};
  const __bf16* wb    = waug + (size_t)n * O_AUG * 256;
  const __bf16* brow0 = &xT[(wave * 32 + mrow) * LDS_STRIDE];
  const __bf16* brow1 = brow0 + 16 * LDS_STRIDE;

  // K loop over 256 input channels, 32 per WMMA step.
  for (int kk = 0; kk < 8; ++kk) {
    // 16-bit operand layout (ISA 7.12.2): lanes 0-15 take K = base+{0..7,16..23},
    // lanes 16-31 take base+8 of the same ranges.
    int kb = kk * 32 + (hi ? 8 : 0);
    FragBF bf0, bf1;
    bf0.f[0] = *(const v4f*)(brow0 + kb);        // ds_load_b128
    bf0.f[1] = *(const v4f*)(brow0 + kb + 16);
    bf1.f[0] = *(const v4f*)(brow1 + kb);
    bf1.f[1] = *(const v4f*)(brow1 + kb + 16);
#pragma unroll
    for (int t = 0; t < 17; ++t) {
      const __bf16* ap = wb + (size_t)(t * 16 + mrow) * 256 + kb;
      FragBF afrag;
      afrag.f[0] = *(const v4f*)(ap);            // global_load_b128
      afrag.f[1] = *(const v4f*)(ap + 16);
      acc0[t] = __builtin_amdgcn_wmma_f32_16x16x32_bf16(
          false, afrag.v, false, bf0.v, (short)0, acc0[t], false, false);
      acc1[t] = __builtin_amdgcn_wmma_f32_16x16x32_bf16(
          false, afrag.v, false, bf1.v, (short)0, acc1[t], false, false);
    }
  }

  // Attention logits live in tile 16: VGPR0 = row 256 (u0.x), VGPR1 = row 257
  // (u1.x) for lanes 0-15 (D layout: lane holds N=lane%16, M = r + 8*hi).
  float d0 = dvals[0 * 16 + n];
  float d1 = dvals[1 * 16 + n];

  float a0 = acc0[16][0] + d0;
  float a1 = acc0[16][1] + d1;
  float mxa = fmaxf(a0, a1);
  float ea0 = __expf(a0 - mxa), ea1 = __expf(a1 - mxa);
  float pa0 = ea0 / (ea0 + ea1);
  pa0 = __shfl(pa0, mrow, 32);     // broadcast pixel prob to upper half-lanes
  float pa1 = 1.0f - pa0;

  float c0 = acc1[16][0] + d0;
  float c1 = acc1[16][1] + d1;
  float mxc = fmaxf(c0, c1);
  float ec0 = __expf(c0 - mxc), ec1 = __expf(c1 - mxc);
  float pc0 = ec0 / (ec0 + ec1);
  pc0 = __shfl(pc0, mrow, 32);
  float pc1 = 1.0f - pc0;

  float g = gamma[0];
  int pix = s0 + wave * 32 + mrow;           // N-tile 0; N-tile 1 at +16
  float* outn = out + (size_t)n * 256 * HW + pix;
#pragma unroll
  for (int t = 0; t < 16; ++t) {
#pragma unroll
    for (int r = 0; r < 8; ++r) {
      int o = t * 16 + r + (hi ? 8 : 0);
      float base = s_bf[o];
      float fb = s_bg[o], ff = s_fg[o];
      float v0 = acc0[t][r] + base + g * (pa0 * fb + pa1 * ff);
      float v1 = acc1[t][r] + base + g * (pc0 * fb + pc1 * ff);
      float* op = outn + (size_t)o * HW;
      op[0]  = v0;
      op[16] = v1;
    }
  }
}

// ---------------------------------------------------------------------------
extern "C" void kernel_launch(void* const* d_in, const int* in_sizes, int n_in,
                              void* d_out, int out_size, void* d_ws, size_t ws_size,
                              hipStream_t stream) {
  const float* x     = (const float*)d_in[0];
  const float* bg    = (const float*)d_in[1];
  const float* fg    = (const float*)d_in[2];
  const float* W_fb  = (const float*)d_in[3];
  const float* b_fb  = (const float*)d_in[4];
  const float* W_v   = (const float*)d_in[5];
  const float* b_v   = (const float*)d_in[6];
  const float* W_f   = (const float*)d_in[7];
  const float* b_f   = (const float*)d_in[8];
  const float* gamma = (const float*)d_in[9];
  float* out = (float*)d_out;

  char* ws = (char*)d_ws;
  float*  msum = (float*)(ws);                        // 32 f32
  float*  xm   = (float*)(ws + 256);                  // 2*4096 f32
  float*  feat = (float*)(ws + 256 + 32768);          // 2*4096 f32
  float*  u    = (float*)(ws + 256 + 65536);          // 2*4096 f32
  float*  dv   = (float*)(ws + 256 + 98304);          // 32 f32
  __bf16* waug = (__bf16*)(ws + 256 + 98304 + 256);   // 16*272*256 bf16 (~2.13 MB)

  k_msum<<<32, 256, 0, stream>>>(bg, fg, msum);
  k_xm<<<4096, 256, 0, stream>>>(x, bg, fg, xm);
  k_feat<<<32, 256, 0, stream>>>(W_fb, b_fb, xm, msum, feat);
  k_uvec<<<32, 256, 0, stream>>>(W_v, b_v, feat, u, dv);
  k_build_waug<<<1088, 1024, 0, stream>>>(W_f, u, waug);
  k_main<<<dim3(36, 16), 256, 0, stream>>>(x, waug, b_f, feat, dv, gamma, out);
}
